// SelfAttention_87136296501926
// MI455X (gfx1250) — matrix operand order repaired and checked by reading, not compiled, not run on hardware
//
#include <hip/hip_runtime.h>

// ---------------------------------------------------------------- types
typedef __attribute__((ext_vector_type(16))) __bf16        v16bf;
typedef __attribute__((ext_vector_type(8)))  float         v8f;
typedef __attribute__((ext_vector_type(8)))  unsigned int  u32x8;

#define HN   32      // query heads
#define KHN  8       // kv heads
#define DH   128     // head dim

__device__ __forceinline__ unsigned short f2bf(float f) {
    unsigned int u = __float_as_uint(f);
    unsigned int r = u + 0x7FFFu + ((u >> 16) & 1u);   // round-to-nearest-even
    return (unsigned short)(r >> 16);
}

__device__ __forceinline__ unsigned int pack2bf(float a, float b) {
    return (unsigned int)f2bf(a) | ((unsigned int)f2bf(b) << 16);
}

__device__ __forceinline__ v16bf make_v16bf(uint4 lo, uint4 hi) {
    u32x8 t = {lo.x, lo.y, lo.z, lo.w, hi.x, hi.y, hi.z, hi.w};
    return __builtin_bit_cast(v16bf, t);
}

__device__ __forceinline__ v8f v8f_zero() {
    v8f z = {0.f, 0.f, 0.f, 0.f, 0.f, 0.f, 0.f, 0.f};
    return z;
}

// ---- CDNA5 async memory->LDS copy (tracked by ASYNCcnt), via inline asm ----
__device__ __forceinline__ void async_b128(unsigned lds_off, const void* gaddr) {
    asm volatile("global_load_async_to_lds_b128 %0, %1, off"
                 :: "v"(lds_off), "v"(gaddr) : "memory");
}
__device__ __forceinline__ void wait_async0() {
    asm volatile("s_wait_asynccnt 0x0" ::: "memory");
}
__device__ __forceinline__ unsigned lds_off32(const void* p) {
    // generic LDS address = {shared aperture, 32-bit LDS offset}; truncation
    // yields the LDS byte offset the async instructions expect.
    return (unsigned)(unsigned long long)p;
}

// ---------------------------------------------------------------- fp32 -> bf16 cast (vectorized, n % 4 == 0)
__global__ void cast_f32_to_bf16_v4(const float4* __restrict__ in,
                                    uint2* __restrict__ out, long long n4) {
    long long stride = (long long)gridDim.x * blockDim.x;
    for (long long i = (long long)blockIdx.x * blockDim.x + threadIdx.x; i < n4; i += stride) {
        float4 f = in[i];
        uint2 o;
        o.x = pack2bf(f.x, f.y);
        o.y = pack2bf(f.z, f.w);
        out[i] = o;
    }
}

// ---------------------------------------------------------------- GEMM  C[m,n] = sum_k A[m,k] * W[n,k]
// A: M x K bf16 row-major;  W: N x K bf16 row-major;  C: M x N fp32.
// 256 threads = 8 waves; block tile 128x128; wave tile 64x32; K-step 32.
// Double-buffered LDS; tiles staged with GLOBAL_LOAD_ASYNC_TO_LDS_B128 so the
// next tile's HBM->LDS DMA overlaps the current tile's WMMA work.
#define BM 128
#define BN 128
#define BK 32

__global__ __launch_bounds__(256) void gemm_nt_bf16(
    const unsigned short* __restrict__ A,
    const unsigned short* __restrict__ W,
    float* __restrict__ C, int M, int N, int K)
{
    __shared__ __align__(16) unsigned short As[2][BM][BK];
    __shared__ __align__(16) unsigned short Ws[2][BN][BK];

    const int tid  = threadIdx.x;
    const int lane = tid & 31;
    const int wv   = tid >> 5;
    const int wvM  = wv >> 2;          // 0..1  -> 64-row slab
    const int wvN  = wv & 3;           // 0..3  -> 32-col slab
    const int half = lane >> 4;        // lane 16..31 -> second K/M half
    const int ln   = lane & 15;

    const int m0 = blockIdx.y * BM;
    const int n0 = blockIdx.x * BN;

    // staging assignment: thread owns rows r0 and r0+64, 16B segment sg
    const int r0 = tid >> 2;
    const int sg = (tid & 3) * 8;

    v8f acc[4][2];
    #pragma unroll
    for (int i = 0; i < 4; ++i)
        #pragma unroll
        for (int j = 0; j < 2; ++j) acc[i][j] = v8f_zero();

    auto stage = [&](int buf, int k0) {
        async_b128(lds_off32(&As[buf][r0][sg]),      &A[(size_t)(m0 + r0) * K + k0 + sg]);
        async_b128(lds_off32(&Ws[buf][r0][sg]),      &W[(size_t)(n0 + r0) * K + k0 + sg]);
        async_b128(lds_off32(&As[buf][r0 + 64][sg]), &A[(size_t)(m0 + r0 + 64) * K + k0 + sg]);
        async_b128(lds_off32(&Ws[buf][r0 + 64][sg]), &W[(size_t)(n0 + r0 + 64) * K + k0 + sg]);
    };

    stage(0, 0);
    wait_async0();
    __syncthreads();

    const int nk = K / BK;
    for (int kb = 0; kb < nk; ++kb) {
        const int cur = kb & 1;
        if (kb + 1 < nk) stage(cur ^ 1, (kb + 1) * BK);   // async prefetch of next tile

        v16bf af[4], bfr[2];
        #pragma unroll
        for (int i = 0; i < 4; ++i) {
            int row = wvM * 64 + i * 16 + ln;
            uint4 lo = *(const uint4*)&As[cur][row][8 * half];        // K = 8h..8h+7
            uint4 hi = *(const uint4*)&As[cur][row][16 + 8 * half];   // K = 16+8h..+7
            af[i] = make_v16bf(lo, hi);
        }
        #pragma unroll
        for (int j = 0; j < 2; ++j) {
            int col = wvN * 32 + j * 16 + ln;                         // lane = B column
            uint4 lo = *(const uint4*)&Ws[cur][col][8 * half];        // K = 8h..8h+7
            uint4 hi = *(const uint4*)&Ws[cur][col][16 + 8 * half];
            bfr[j] = make_v16bf(lo, hi);
        }
        #pragma unroll
        for (int i = 0; i < 4; ++i)
            #pragma unroll
            for (int j = 0; j < 2; ++j)
                acc[i][j] = __builtin_amdgcn_wmma_f32_16x16x32_bf16(
                    false, af[i], false, bfr[j], (short)0, acc[i][j], false, false);

        wait_async0();       // this wave's async DMAs for the next tile complete
        __syncthreads();     // all waves' DMAs + reads of current tile complete
    }

    #pragma unroll
    for (int i = 0; i < 4; ++i)
        #pragma unroll
        for (int j = 0; j < 2; ++j)
            #pragma unroll
            for (int v = 0; v < 8; ++v) {
                int r  = m0 + wvM * 64 + i * 16 + v + 8 * half;
                int cc = n0 + wvN * 32 + j * 16 + ln;
                C[(size_t)r * N + cc] = acc[i][j][v];
            }
}

// ---------------------------------------------------------------- RoPE on Q -> bf16
__global__ void rope_q_kernel(const float* __restrict__ q,
                              const float* __restrict__ freqs,
                              unsigned short* __restrict__ qr,
                              int S, long long total)
{
    long long stride = (long long)gridDim.x * blockDim.x;
    for (long long i = (long long)blockIdx.x * blockDim.x + threadIdx.x; i < total; i += stride) {
        int d2  = (int)(i & 63);
        int hh  = (int)((i >> 6) & 31);
        int row = (int)(i >> 11);
        int s   = row % S;
        float f  = freqs[s * 64 + d2];
        float c  = __cosf(f), sn = __sinf(f);
        size_t base = (size_t)row * 4096 + hh * 128 + d2 * 2;
        float e = q[base], o = q[base + 1];
        qr[base]     = f2bf(e * c - o * sn);
        qr[base + 1] = f2bf(e * sn + o * c);
    }
}

// ---------------------------------------------------------------- RoPE on K + scatter to keys[b][start+s]
__global__ void rope_k_scatter(const float* __restrict__ k,
                               const float* __restrict__ freqs,
                               unsigned short* __restrict__ keys,
                               const int* __restrict__ sp,
                               int S, int MAXS, long long total)
{
    int start = *sp;
    long long stride = (long long)gridDim.x * blockDim.x;
    for (long long i = (long long)blockIdx.x * blockDim.x + threadIdx.x; i < total; i += stride) {
        int d2  = (int)(i & 63);
        int kh  = (int)((i >> 6) & 7);
        int row = (int)(i >> 9);
        int b = row / S, s = row % S;
        float f  = freqs[s * 64 + d2];
        float c  = __cosf(f), sn = __sinf(f);
        int col = kh * 128 + d2 * 2;
        size_t src = (size_t)row * 1024 + col;
        size_t dst = ((size_t)b * MAXS + start + s) * 1024 + col;
        float e = k[src], o = k[src + 1];
        keys[dst]     = f2bf(e * c - o * sn);
        keys[dst + 1] = f2bf(e * sn + o * c);
    }
}

// ---------------------------------------------------------------- V scatter (no rope)
__global__ void v_scatter(const float* __restrict__ v,
                          unsigned short* __restrict__ vals,
                          const int* __restrict__ sp,
                          int S, int MAXS, long long total)  // total = M*1024
{
    int start = *sp;
    long long stride = (long long)gridDim.x * blockDim.x;
    for (long long i = (long long)blockIdx.x * blockDim.x + threadIdx.x; i < total; i += stride) {
        int col = (int)(i & 1023);
        int row = (int)(i >> 10);
        int b = row / S, s = row % S;
        vals[((size_t)b * MAXS + start + s) * 1024 + col] = f2bf(v[(size_t)row * 1024 + col]);
    }
}

// ---------------------------------------------------------------- old cache -> bf16 (t < start)
__global__ void cache_cast(const float* __restrict__ cache,
                           unsigned short* __restrict__ out,
                           const int* __restrict__ sp,
                           int MAXS, long long total)   // total = B*MAXS*1024
{
    int start = *sp;
    long long stride = (long long)gridDim.x * blockDim.x;
    for (long long i = (long long)blockIdx.x * blockDim.x + threadIdx.x; i < total; i += stride) {
        int t = (int)((i >> 10) % MAXS);
        if (t < start) out[i] = f2bf(cache[i]);
    }
}

// ---------------------------------------------------------------- flash attention
// 128 threads = 4 waves; each wave: 16 queries x full D=128; key blocks of 32.
// K blocks staged with async-to-LDS; V transposed through VGPRs on the fly.
__global__ __launch_bounds__(128) void attn_kernel(
    const unsigned short* __restrict__ qr,    // (B,S,H,D) bf16
    const unsigned short* __restrict__ keys,  // (B,MAXS,KH,D) bf16
    const unsigned short* __restrict__ vals,  // (B,MAXS,KH,D) bf16
    unsigned short* __restrict__ ao,          // (B,S,H*D) bf16
    const int* __restrict__ sp, int S, int MAXS)
{
    __shared__ __align__(16) unsigned short Qs[64][DH];   // 16 KB
    __shared__ __align__(16) unsigned short Ks[32][DH];   // 8 KB
    __shared__ __align__(16) unsigned short Vt[DH][32];   // 8 KB (V transposed)
    __shared__ __align__(16) unsigned short Ps[4][16][32];// 4 KB (per-wave P patch)

    const int tid  = threadIdx.x;
    const int lane = tid & 31;
    const int wv   = tid >> 5;
    const int half = lane >> 4;
    const int ln   = lane & 15;

    const int bh = blockIdx.y;
    const int b  = bh / HN;
    const int h  = bh % HN;
    const int kh = h >> 2;                 // H/KH = 4
    const int s0 = blockIdx.x * 64;
    const int T  = *sp + S;

    // stage Q tile (64 x 128)
    for (int u = tid; u < 64 * DH / 8; u += 128) {
        int r = u >> 4, sgq = (u & 15) * 8;
        *(uint4*)&Qs[r][sgq] =
            *(const uint4*)&qr[(((size_t)b * S + s0 + r) * HN + h) * DH + sgq];
    }

    v8f o[8];
    #pragma unroll
    for (int j = 0; j < 8; ++j) o[j] = v8f_zero();
    float mrow[8], lrow[8];
    #pragma unroll
    for (int v = 0; v < 8; ++v) { mrow[v] = -1e30f; lrow[v] = 0.f; }

    const int qrowA = wv * 16 + ln;        // A-fragment row for this lane
    const int nb = (T + 31) >> 5;

    for (int kb = 0; kb < nb; ++kb) {
        const int t0 = kb * 32;
        __syncthreads();                   // protect Ks/Vt reuse
        #pragma unroll
        for (int u = tid; u < 512; u += 128) {             // K block via async DMA
            int r = u >> 4, sgk = (u & 15) * 8;
            async_b128(lds_off32(&Ks[r][sgk]),
                       &keys[(((size_t)b * MAXS + t0 + r) * KHN + kh) * DH + sgk]);
        }
        #pragma unroll
        for (int u = tid; u < 512; u += 128) {             // V block, transposed into Vt
            int r = u >> 4, sgv = (u & 15) * 8;
            uint4 pv = *(const uint4*)&vals[(((size_t)b * MAXS + t0 + r) * KHN + kh) * DH + sgv];
            const unsigned short* pw = (const unsigned short*)&pv;
            #pragma unroll
            for (int j = 0; j < 8; ++j) Vt[sgv + j][r] = pw[j];
        }
        wait_async0();
        __syncthreads();

        // ---- scores S = Q . K^T   (16 x 32), K-dim = 128 in 4 steps
        v8f sc[2];
        sc[0] = v8f_zero(); sc[1] = v8f_zero();
        #pragma unroll
        for (int dk = 0; dk < 4; ++dk) {
            uint4 alo = *(const uint4*)&Qs[qrowA][dk * 32 + 8 * half];
            uint4 ahi = *(const uint4*)&Qs[qrowA][dk * 32 + 16 + 8 * half];
            v16bf aq = make_v16bf(alo, ahi);
            #pragma unroll
            for (int t = 0; t < 2; ++t) {
                int kcol = t * 16 + ln;
                uint4 blo = *(const uint4*)&Ks[kcol][dk * 32 + 8 * half];
                uint4 bhi = *(const uint4*)&Ks[kcol][dk * 32 + 16 + 8 * half];
                v16bf bk = make_v16bf(blo, bhi);
                sc[t] = __builtin_amdgcn_wmma_f32_16x16x32_bf16(
                    false, aq, false, bk, (short)0, sc[t], false, false);
            }
        }

        // ---- online softmax, write P (bf16) into wave-private LDS patch
        const float scal = 0.08838834764831845f;   // 1/sqrt(128)
        #pragma unroll
        for (int v = 0; v < 8; ++v) {
            float x0 = sc[0][v] * scal;
            float x1 = sc[1][v] * scal;
            if (t0 + ln      >= T) x0 = -1e30f;
            if (t0 + 16 + ln >= T) x1 = -1e30f;
            float mx = fmaxf(x0, x1);
            #pragma unroll
            for (int d = 1; d < 16; d <<= 1) mx = fmaxf(mx, __shfl_xor(mx, d, 16));
            float mnew = fmaxf(mrow[v], mx);
            float corr = __expf(mrow[v] - mnew);
            float p0 = __expf(x0 - mnew);
            float p1 = __expf(x1 - mnew);
            float sum = p0 + p1;
            #pragma unroll
            for (int d = 1; d < 16; d <<= 1) sum += __shfl_xor(sum, d, 16);
            lrow[v] = lrow[v] * corr + sum;
            mrow[v] = mnew;
            #pragma unroll
            for (int j = 0; j < 8; ++j) o[j][v] *= corr;
            Ps[wv][v + 8 * half][ln]      = f2bf(p0);
            Ps[wv][v + 8 * half][16 + ln] = f2bf(p1);
        }

        // ---- O += P . V   (same-wave LDS writes/reads stay in order)
        uint4 plo = *(const uint4*)&Ps[wv][ln][8 * half];
        uint4 phi = *(const uint4*)&Ps[wv][ln][16 + 8 * half];
        v16bf pa = make_v16bf(plo, phi);
        #pragma unroll
        for (int j = 0; j < 8; ++j) {
            int dcol = j * 16 + ln;
            uint4 vlo = *(const uint4*)&Vt[dcol][8 * half];
            uint4 vhi = *(const uint4*)&Vt[dcol][16 + 8 * half];
            v16bf vb = make_v16bf(vlo, vhi);
            o[j] = __builtin_amdgcn_wmma_f32_16x16x32_bf16(
                false, pa, false, vb, (short)0, o[j], false, false);
        }
    }

    // ---- epilogue: normalize + store bf16
    #pragma unroll
    for (int v = 0; v < 8; ++v) {
        float inv = 1.0f / lrow[v];
        int q = s0 + wv * 16 + v + 8 * half;
        size_t base = ((size_t)b * S + q) * 4096 + h * DH;
        #pragma unroll
        for (int j = 0; j < 8; ++j)
            ao[base + j * 16 + ln] = f2bf(o[j][v] * inv);
    }
}

// ---------------------------------------------------------------- launcher
extern "C" void kernel_launch(void* const* d_in, const int* in_sizes, int n_in,
                              void* d_out, int out_size, void* d_ws, size_t ws_size,
                              hipStream_t stream) {
    const float* x       = (const float*)d_in[0];
    const float* wq      = (const float*)d_in[1];
    const float* wk      = (const float*)d_in[2];
    const float* wvv     = (const float*)d_in[3];
    const float* wo      = (const float*)d_in[4];
    const float* k_cache = (const float*)d_in[5];
    const float* v_cache = (const float*)d_in[6];
    const float* freqs   = (const float*)d_in[7];
    const int*   sp      = (const int*)d_in[8];

    const int D = DH, H = HN, KH = KHN;
    const int E   = H * D;                       // 4096
    const int KVE = KH * D;                      // 1024
    const int S   = in_sizes[7] / (D / 2);       // 1024
    const int B   = in_sizes[0] / (S * E);       // 2
    const int MAXS = in_sizes[5] / (B * KVE);    // 2048
    const int M = B * S;                         // 2048

    char* w = (char*)d_ws;
    auto bump = [&](size_t bytes) {
        void* p = (void*)w;
        w += (bytes + 255) & ~(size_t)255;
        return p;
    };
    unsigned short* xb    = (unsigned short*)bump((size_t)M * E * 2);
    unsigned short* wqb   = (unsigned short*)bump((size_t)E * E * 2);
    unsigned short* wkb   = (unsigned short*)bump((size_t)KVE * E * 2);
    unsigned short* wvb   = (unsigned short*)bump((size_t)KVE * E * 2);
    unsigned short* wob   = (unsigned short*)bump((size_t)E * E * 2);
    float*          qf    = (float*)bump((size_t)M * E * 4);
    float*          kf    = (float*)bump((size_t)M * KVE * 4);
    float*          vf    = (float*)bump((size_t)M * KVE * 4);
    unsigned short* qrope = (unsigned short*)bump((size_t)M * E * 2);
    unsigned short* keysb = (unsigned short*)bump((size_t)B * MAXS * KVE * 2);
    unsigned short* valsb = (unsigned short*)bump((size_t)B * MAXS * KVE * 2);
    unsigned short* aob   = (unsigned short*)bump((size_t)M * E * 2);

    auto gblocks = [](long long n) {
        long long b = (n + 255) / 256;
        return (int)(b > 65535 ? 65535 : b);
    };

    // 1) casts to bf16 (vectorized x4)
    cast_f32_to_bf16_v4<<<gblocks((long long)M * E / 4),   256, 0, stream>>>(
        (const float4*)x,   (uint2*)xb,  (long long)M * E / 4);
    cast_f32_to_bf16_v4<<<gblocks((long long)E * E / 4),   256, 0, stream>>>(
        (const float4*)wq,  (uint2*)wqb, (long long)E * E / 4);
    cast_f32_to_bf16_v4<<<gblocks((long long)KVE * E / 4), 256, 0, stream>>>(
        (const float4*)wk,  (uint2*)wkb, (long long)KVE * E / 4);
    cast_f32_to_bf16_v4<<<gblocks((long long)KVE * E / 4), 256, 0, stream>>>(
        (const float4*)wvv, (uint2*)wvb, (long long)KVE * E / 4);
    cast_f32_to_bf16_v4<<<gblocks((long long)E * E / 4),   256, 0, stream>>>(
        (const float4*)wo,  (uint2*)wob, (long long)E * E / 4);

    // 2) projections (WMMA GEMMs, async double-buffered staging)
    gemm_nt_bf16<<<dim3(E / BN,   M / BM), 256, 0, stream>>>(xb, wqb, qf, M, E,   E);
    gemm_nt_bf16<<<dim3(KVE / BN, M / BM), 256, 0, stream>>>(xb, wkb, kf, M, KVE, E);
    gemm_nt_bf16<<<dim3(KVE / BN, M / BM), 256, 0, stream>>>(xb, wvb, vf, M, KVE, E);

    // 3) RoPE + cache assembly (start_pos read on-device; capture-safe)
    rope_q_kernel<<<gblocks((long long)M * H * (D / 2)), 256, 0, stream>>>(
        qf, freqs, qrope, S, (long long)M * H * (D / 2));
    rope_k_scatter<<<gblocks((long long)M * KH * (D / 2)), 256, 0, stream>>>(
        kf, freqs, keysb, sp, S, MAXS, (long long)M * KH * (D / 2));
    v_scatter<<<gblocks((long long)M * KVE), 256, 0, stream>>>(
        vf, valsb, sp, S, MAXS, (long long)M * KVE);
    cache_cast<<<gblocks((long long)B * MAXS * KVE), 256, 0, stream>>>(
        k_cache, keysb, sp, MAXS, (long long)B * MAXS * KVE);
    cache_cast<<<gblocks((long long)B * MAXS * KVE), 256, 0, stream>>>(
        v_cache, valsb, sp, MAXS, (long long)B * MAXS * KVE);

    // 4) flash attention (WMMA + async K staging)
    attn_kernel<<<dim3(S / 64, B * H), 128, 0, stream>>>(
        qrope, keysb, valsb, aob, sp, S, MAXS);

    // 5) output projection -> fp32 d_out
    gemm_nt_bf16<<<dim3(E / BN, M / BM), 256, 0, stream>>>(
        aob, wob, (float*)d_out, M, E, E);
}
